// GlobalPath_24438363914545
// MI455X (gfx1250) — compile-verified, matrix operand order checked by reading, not run on hardware
//
#include <hip/hip_runtime.h>
#include <hip/hip_bf16.h>

#define HH    48
#define WW    48
#define L_TOK 2304        // 48*48
#define NB    4
#define NTOK  9216        // NB * L_TOK
#define C_IN  256
#define CEX   512
#define NSTATE 16
#define DTR   32

typedef __attribute__((ext_vector_type(16))) __bf16 v16bf;
typedef __attribute__((ext_vector_type(8)))  __bf16 v8bf;
typedef __attribute__((ext_vector_type(8)))  float  v8f;
typedef __attribute__((ext_vector_type(4)))  unsigned int u32x4;
typedef __attribute__((ext_vector_type(8)))  int i32x8;
typedef __attribute__((ext_vector_type(4)))  int i32x4;

__device__ __forceinline__ __bf16 f2bf(float f) {
  unsigned u = __builtin_bit_cast(unsigned, f);
  unsigned r = (u + 0x7FFFu + ((u >> 16) & 1u)) >> 16;
  unsigned short s = (unsigned short)r;
  return __builtin_bit_cast(__bf16, s);
}

// Load one 16x32 bf16 fragment from a row-major tile with leading dim ldk.
// CDNA5 16-bit A layout: lanes 0-15: row=lane, K = k0+0..7 / k0+16..23;
// lanes 16-31: row=lane-16, K = k0+8..15 / k0+24..31.
__device__ __forceinline__ v16bf load_frag16x32(const __bf16* base, int ldk, int k0, int lane) {
  int r    = lane & 15;
  int koff = (lane >> 4) << 3;
  const __bf16* p = base + (size_t)r * ldk + k0 + koff;
  union { v16bf v; v8bf h[2]; } u;
  u.h[0] = *reinterpret_cast<const v8bf*>(p);
  u.h[1] = *reinterpret_cast<const v8bf*>(p + 16);
  return u.v;
}

// ---------------- weight conversion ----------------
__global__ void k_f32_to_bf16(const float* __restrict__ src, __bf16* __restrict__ dst, int n) {
  int i = blockIdx.x * blockDim.x + threadIdx.x;
  if (i < n) dst[i] = f2bf(src[i]);
}

// ---------------- LayerNorm over C (NCHW input) ----------------
__global__ void k_layernorm(const float* __restrict__ x, const float* __restrict__ w,
                            const float* __restrict__ b, __bf16* __restrict__ xn) {
  int gid = blockIdx.x * blockDim.x + threadIdx.x;
  if (gid >= NTOK) return;
  int bidx = gid / L_TOK;
  int hw   = gid % L_TOK;
  const float* xb = x + (size_t)bidx * C_IN * L_TOK + hw;
  float s = 0.f, ss = 0.f;
  for (int c = 0; c < C_IN; ++c) {
    float v = xb[(size_t)c * L_TOK];
    s += v; ss += v * v;
  }
  float mu  = s * (1.f / C_IN);
  float var = ss * (1.f / C_IN) - mu * mu;
  float rs  = rsqrtf(var + 1e-6f);
  __bf16* o = xn + (size_t)gid * C_IN;
  for (int c = 0; c < C_IN; ++c) {
    float v = (xb[(size_t)c * L_TOK] - mu) * rs * w[c] + b[c];
    o[c] = f2bf(v);
  }
}

// ---------------- GEMM 1: expand  uv = xn @ w_ex^T + b_ex ----------------
// M=9216 K=256 N=1024. Workgroup = 8 waves = 8 M-tiles x 1 shared N-tile.
// The 16x256 bf16 W-tile (8KB) is staged into LDS once per block via the
// Tensor Data Mover (tensor_load_to_lds, TENSORcnt), then all 8 waves read
// B-fragments from LDS (ds_load_b128) -> 8x less L2 traffic for W.
__global__ void k_gemm_expand_tdm(const __bf16* __restrict__ A, const __bf16* __restrict__ W,
                                  const float* __restrict__ bias,
                                  float* __restrict__ U, float* __restrict__ V) {
  __shared__ __bf16 sW[16 * C_IN];   // 8 KB
  const int tiles_n = 1024 / 16;
  int tn  = blockIdx.x % tiles_n;
  int tm8 = blockIdx.x / tiles_n;    // super-tile of 8 M-tiles
  int wid  = threadIdx.x >> 5;
  int lane = threadIdx.x & 31;

  // --- TDM: stage W[tn*16 .. tn*16+15][0..255] into LDS (wave 0 issues) ---
  if (wid == 0) {
    unsigned long long ga = (unsigned long long)(const void*)(W + (size_t)tn * 16 * C_IN);
    unsigned lds_off = (unsigned)(size_t)(&sW[0]);   // low 32 bits = LDS byte offset
    // D# group 0: count=1 | lds_addr | global_addr[56:0] | type=2
    u32x4 g0;
    g0.x = 1u;                                   // count=1, user descriptor
    g0.y = lds_off;                              // lds_addr (bytes)
    g0.z = (unsigned)(ga & 0xFFFFFFFFull);       // global_addr[31:0]
    g0.w = (unsigned)((ga >> 32) & 0x01FFFFFFull) | (2u << 30); // ga[56:32] | type=2
    // D# group 1: data_size=1 (2B), tensor_dim0=256, tensor_dim1=1024,
    // tile_dim0=256, tile_dim1=16, tensor_dim0_stride=256
    i32x8 g1;
    g1[0] = (1 << 16);                 // workgroup_mask=0, data_size=1 (2 bytes)
    g1[1] = (int)((C_IN & 0xFFFF) << 16);        // tensor_dim0 lo16 at [63:48]
    g1[2] = (int)(((C_IN >> 16) & 0xFFFF) | ((1024 & 0xFFFF) << 16)); // dim0 hi | dim1 lo
    g1[3] = (int)(((1024 >> 16) & 0xFFFF) | ((C_IN & 0xFFFF) << 16)); // dim1 hi | tile_dim0
    g1[4] = 16;                        // tile_dim1=16, tile_dim2=0
    g1[5] = C_IN;                      // tensor_dim0_stride lo32
    g1[6] = 0;                         // stride hi | tensor_dim1_stride lo
    g1[7] = 0;
    i32x4 g2 = {0, 0, 0, 0};
    i32x4 g3 = {0, 0, 0, 0};
#if defined(__clang_major__) && (__clang_major__ >= 23)
    i32x8 g4 = {0, 0, 0, 0, 0, 0, 0, 0};
    __builtin_amdgcn_tensor_load_to_lds(g0, g1, g2, g3, g4, 0);
#else
    __builtin_amdgcn_tensor_load_to_lds(g0, g1, g2, g3, 0);
#endif
    __builtin_amdgcn_s_wait_tensorcnt(0);
  }
  __syncthreads();

  int tm = tm8 * 8 + wid;
  const __bf16* Ab = A + (size_t)tm * 16 * C_IN;
  int r    = lane & 15;
  int koff = (lane >> 4) << 3;
  v8f acc = {};
#pragma unroll
  for (int k = 0; k < C_IN; k += 32) {
    if (k + 32 < C_IN)
      __builtin_prefetch(Ab + (size_t)r * C_IN + k + 32, 0, 0);
    v16bf a = load_frag16x32(Ab, C_IN, k, lane);
    union { v16bf v; v8bf h[2]; } wu;
    wu.h[0] = *reinterpret_cast<const v8bf*>(&sW[r * C_IN + k + koff]);
    wu.h[1] = *reinterpret_cast<const v8bf*>(&sW[r * C_IN + k + koff + 16]);
    acc = __builtin_amdgcn_wmma_f32_16x16x32_bf16(false, a, false, wu.v, (short)0, acc, false, false);
  }
  int n = tn * 16 + (lane & 15);
  float bn = bias[n];
  int mh = (lane >> 4) * 8;
#pragma unroll
  for (int i = 0; i < 8; ++i) {
    int m = tm * 16 + mh + i;
    float val = acc[i] + bn;
    if (n < CEX) U[(size_t)m * CEX + n] = val;
    else         V[(size_t)m * CEX + (n - CEX)] = val;
  }
}

// ---------------- depthwise 3x3 conv + SiLU ----------------
__global__ void k_dwconv_silu(const float* __restrict__ V, const float* __restrict__ kern,
                              float* __restrict__ VA, __bf16* __restrict__ VAB) {
  size_t idx = (size_t)blockIdx.x * blockDim.x + threadIdx.x;
  if (idx >= (size_t)NTOK * CEX) return;
  int c      = (int)(idx % CEX);
  size_t tok = idx / CEX;
  int hw = (int)(tok % L_TOK);
  int bb = (int)(tok / L_TOK);
  int h = hw / WW, w = hw % WW;
  float acc = 0.f;
#pragma unroll
  for (int kh = 0; kh < 3; ++kh) {
    int hh = h + kh - 1;
    if (hh < 0 || hh >= HH) continue;
#pragma unroll
    for (int kw = 0; kw < 3; ++kw) {
      int ww2 = w + kw - 1;
      if (ww2 < 0 || ww2 >= WW) continue;
      size_t t2 = (size_t)bb * L_TOK + hh * WW + ww2;
      acc += V[t2 * CEX + c] * kern[(kh * 3 + kw) * CEX + c];
    }
  }
  float s = acc / (1.f + __expf(-acc));
  VA[idx]  = s;
  VAB[idx] = f2bf(s);
}

// ---------------- GEMM 2: x-proj  dbc = v @ W_xproj^T + b ----------------
__global__ void k_gemm_xproj(const __bf16* __restrict__ A, const __bf16* __restrict__ W,
                             const float* __restrict__ bias,
                             float* __restrict__ DBC, __bf16* __restrict__ DTb) {
  const int tiles_n = 64 / 16;
  int wid  = (blockIdx.x * blockDim.x + threadIdx.x) >> 5;
  int lane = threadIdx.x & 31;
  if (wid >= (NTOK / 16) * tiles_n) return;
  int tm = wid / tiles_n, tn = wid % tiles_n;
  const __bf16* Ab = A + (size_t)tm * 16 * CEX;
  const __bf16* Wb = W + (size_t)tn * 16 * CEX;
  v8f acc = {};
#pragma unroll
  for (int k = 0; k < CEX; k += 32) {
    v16bf a = load_frag16x32(Ab, CEX, k, lane);
    v16bf w = load_frag16x32(Wb, CEX, k, lane);
    acc = __builtin_amdgcn_wmma_f32_16x16x32_bf16(false, a, false, w, (short)0, acc, false, false);
  }
  int n = tn * 16 + (lane & 15);
  float bn = bias[n];
  int mh = (lane >> 4) * 8;
#pragma unroll
  for (int i = 0; i < 8; ++i) {
    int m = tm * 16 + mh + i;
    float val = acc[i] + bn;
    DBC[(size_t)m * 64 + n] = val;
    if (n < DTR) DTb[(size_t)m * DTR + n] = f2bf(val);
  }
}

// ---------------- GEMM 3: delta = softplus(dt @ W_dt^T + b_dt) ----------------
__global__ void k_gemm_dt(const __bf16* __restrict__ A, const __bf16* __restrict__ W,
                          const float* __restrict__ bias, float* __restrict__ DLT) {
  const int tiles_n = CEX / 16;
  int wid  = (blockIdx.x * blockDim.x + threadIdx.x) >> 5;
  int lane = threadIdx.x & 31;
  if (wid >= (NTOK / 16) * tiles_n) return;
  int tm = wid / tiles_n, tn = wid % tiles_n;
  v16bf a = load_frag16x32(A + (size_t)tm * 16 * DTR, DTR, 0, lane);
  v16bf w = load_frag16x32(W + (size_t)tn * 16 * DTR, DTR, 0, lane);
  v8f acc = {};
  acc = __builtin_amdgcn_wmma_f32_16x16x32_bf16(false, a, false, w, (short)0, acc, false, false);
  int n = tn * 16 + (lane & 15);
  float bn = bias[n];
  int mh = (lane >> 4) * 8;
#pragma unroll
  for (int i = 0; i < 8; ++i) {
    int m = tm * 16 + mh + i;
    float z = acc[i] + bn;
    float sp = (z > 20.f) ? z : log1pf(__expf(z));
    DLT[(size_t)m * CEX + n] = sp;
  }
}

// ---------------- selective scan (sequential over L) ----------------
__global__ void k_scan(const float* __restrict__ DBC, const float* __restrict__ DLT,
                       const float* __restrict__ VA, const float* __restrict__ A_log,
                       const float* __restrict__ Dv, float* __restrict__ Y) {
  int c  = blockIdx.x * blockDim.x + threadIdx.x;   // 0..511
  int bb = blockIdx.y;                              // batch
  __shared__ float sB[NSTATE], sC[NSTATE];
  float An[NSTATE];
#pragma unroll
  for (int n = 0; n < NSTATE; ++n)
    An[n] = -__expf(A_log[(size_t)c * NSTATE + n]) * 1.4426950408889634f;
  float Dc = Dv[c];
  float h[NSTATE];
#pragma unroll
  for (int n = 0; n < NSTATE; ++n) h[n] = 0.f;

  for (int t = 0; t < L_TOK; ++t) {
    size_t tok = (size_t)bb * L_TOK + t;
    if (threadIdx.x < NSTATE)
      sB[threadIdx.x] = DBC[tok * 64 + DTR + threadIdx.x];
    else if (threadIdx.x < 2 * NSTATE)
      sC[threadIdx.x - NSTATE] = DBC[tok * 64 + DTR + NSTATE + (threadIdx.x - NSTATE)];
    __syncthreads();
    float d  = DLT[tok * CEX + c];
    float v  = VA[tok * CEX + c];
    float dv = d * v;
    float y  = 0.f;
#pragma unroll
    for (int n = 0; n < NSTATE; ++n) {
      float dA = exp2f(d * An[n]);
      h[n] = h[n] * dA + dv * sB[n];
      y += h[n] * sC[n];
    }
    Y[tok * CEX + c] = y + Dc * v;
    __syncthreads();
  }
}

// ---------------- combine: ycomb = y*silu(u) + v_act*u*sigmoid(-u) ----------------
__global__ void k_combine(const float* __restrict__ Y, const float* __restrict__ U,
                          const float* __restrict__ VA, __bf16* __restrict__ YC) {
  size_t idx = (size_t)blockIdx.x * blockDim.x + threadIdx.x;
  if (idx >= (size_t)NTOK * CEX) return;
  float u = U[idx], y = Y[idx], va = VA[idx];
  float sig = 1.f / (1.f + __expf(-u));
  float g    = u * sig;
  float gate = va * u * (1.f - sig);
  YC[idx] = f2bf(y * g + gate);
}

// ---------------- GEMM 4: out = ycomb @ w_proj^T + b_proj, NCHW scatter ----------------
__global__ void k_gemm_proj(const __bf16* __restrict__ A, const __bf16* __restrict__ W,
                            const float* __restrict__ bias, float* __restrict__ out) {
  const int tiles_n = C_IN / 16;
  int wid  = (blockIdx.x * blockDim.x + threadIdx.x) >> 5;
  int lane = threadIdx.x & 31;
  if (wid >= (NTOK / 16) * tiles_n) return;
  int tm = wid / tiles_n, tn = wid % tiles_n;
  const __bf16* Ab = A + (size_t)tm * 16 * CEX;
  const __bf16* Wb = W + (size_t)tn * 16 * CEX;
  v8f acc = {};
#pragma unroll
  for (int k = 0; k < CEX; k += 32) {
    v16bf a = load_frag16x32(Ab, CEX, k, lane);
    v16bf w = load_frag16x32(Wb, CEX, k, lane);
    acc = __builtin_amdgcn_wmma_f32_16x16x32_bf16(false, a, false, w, (short)0, acc, false, false);
  }
  int n = tn * 16 + (lane & 15);
  float bn = bias[n];
  int mh = (lane >> 4) * 8;
#pragma unroll
  for (int i = 0; i < 8; ++i) {
    int m  = tm * 16 + mh + i;
    int bb = m / L_TOK, hw = m % L_TOK;
    out[((size_t)bb * C_IN + n) * L_TOK + hw] = acc[i] + bn;
  }
}

extern "C" void kernel_launch(void* const* d_in, const int* in_sizes, int n_in,
                              void* d_out, int out_size, void* d_ws, size_t ws_size,
                              hipStream_t stream) {
  const float* x      = (const float*)d_in[0];
  const float* norm_w = (const float*)d_in[1];
  const float* norm_b = (const float*)d_in[2];
  const float* w_ex   = (const float*)d_in[3];
  const float* b_ex   = (const float*)d_in[4];
  const float* w_proj = (const float*)d_in[5];
  const float* b_proj = (const float*)d_in[6];
  const float* dw_k   = (const float*)d_in[7];
  const float* A_log  = (const float*)d_in[8];
  const float* Dvec   = (const float*)d_in[9];
  const float* W_xp   = (const float*)d_in[10];
  const float* b_xp   = (const float*)d_in[11];
  const float* W_dt   = (const float*)d_in[12];
  const float* b_dt   = (const float*)d_in[13];
  float* out = (float*)d_out;

  char* ws = (char*)d_ws;
  size_t off = 0;
  auto carve = [&](size_t bytes) { char* p = ws + off; off = (off + bytes + 255) & ~(size_t)255; return p; };
  __bf16* XN   = (__bf16*)carve((size_t)NTOK * C_IN * 2);
  __bf16* WEXb = (__bf16*)carve((size_t)2 * CEX * C_IN * 2);
  __bf16* WXPb = (__bf16*)carve((size_t)64 * CEX * 2);
  __bf16* WDTb = (__bf16*)carve((size_t)CEX * DTR * 2);
  __bf16* WPJb = (__bf16*)carve((size_t)C_IN * CEX * 2);
  float*  Ubuf = (float*)carve((size_t)NTOK * CEX * 4);
  float*  Vbuf = (float*)carve((size_t)NTOK * CEX * 4);   // pre-conv v; reused as Y
  float*  VA   = (float*)carve((size_t)NTOK * CEX * 4);
  __bf16* VAB  = (__bf16*)carve((size_t)NTOK * CEX * 2);
  float*  DBC  = (float*)carve((size_t)NTOK * 64 * 4);
  __bf16* DTb  = (__bf16*)carve((size_t)NTOK * DTR * 2);
  float*  DLT  = (float*)carve((size_t)NTOK * CEX * 4);
  __bf16* YC   = (__bf16*)carve((size_t)NTOK * CEX * 2);
  float*  Y    = Vbuf;

  const int TPB = 256;
  k_f32_to_bf16<<<(2 * CEX * C_IN + TPB - 1) / TPB, TPB, 0, stream>>>(w_ex, WEXb, 2 * CEX * C_IN);
  k_f32_to_bf16<<<(64 * CEX + TPB - 1) / TPB, TPB, 0, stream>>>(W_xp, WXPb, 64 * CEX);
  k_f32_to_bf16<<<(CEX * DTR + TPB - 1) / TPB, TPB, 0, stream>>>(W_dt, WDTb, CEX * DTR);
  k_f32_to_bf16<<<(C_IN * CEX + TPB - 1) / TPB, TPB, 0, stream>>>(w_proj, WPJb, C_IN * CEX);

  k_layernorm<<<(NTOK + TPB - 1) / TPB, TPB, 0, stream>>>(x, norm_w, norm_b, XN);

  // expand GEMM: 72 M-supertiles x 64 N-tiles, W staged via TDM
  k_gemm_expand_tdm<<<(NTOK / 128) * (1024 / 16), TPB, 0, stream>>>(XN, WEXb, b_ex, Ubuf, Vbuf);

  k_dwconv_silu<<<((size_t)NTOK * CEX + TPB - 1) / TPB, TPB, 0, stream>>>(Vbuf, dw_k, VA, VAB);

  k_gemm_xproj<<<(NTOK / 16) * (64 / 16) / 8, TPB, 0, stream>>>(VAB, WXPb, b_xp, DBC, DTb);

  k_gemm_dt<<<(NTOK / 16) * (CEX / 16) / 8, TPB, 0, stream>>>(DTb, WDTb, b_dt, DLT);

  k_scan<<<dim3(CEX / TPB, NB), TPB, 0, stream>>>(DBC, DLT, VA, A_log, Dvec, Y);

  k_combine<<<((size_t)NTOK * CEX + TPB - 1) / TPB, TPB, 0, stream>>>(Y, Ubuf, VA, YC);

  k_gemm_proj<<<(NTOK / 16) * (C_IN / 16) / 8, TPB, 0, stream>>>(YC, WPJb, b_proj, out);
  (void)in_sizes; (void)n_in; (void)out_size; (void)ws_size;
}
